// Sequence_77902116815035
// MI455X (gfx1250) — matrix operand order repaired and checked by reading, not compiled, not run on hardware
//
#include <hip/hip_runtime.h>
#include <hip/hip_bf16.h>

typedef __attribute__((ext_vector_type(16))) _Float16 v16h;
typedef __attribute__((ext_vector_type(8)))  float    v8f;

#define IN_DIM 6
#define H_DIM  32
#define T_LEN  1024
#define B_TOT  4096
#define NTILE  8   // 128 gate columns / 16

union U16h {
  uint4    u[2];
  v16h     v;
  _Float16 h[16];
};

// Single-instruction hardware tanh (CDNA5 TRANS unit).
__device__ __forceinline__ float fast_tanh(float x) {
#if __has_builtin(__builtin_amdgcn_tanhf)
  return __builtin_amdgcn_tanhf(x);
#else
  float r;
  // v_nop satisfies the TRANS-op "1 instruction before result use" hazard rule,
  // since the compiler cannot model latency of inline asm.
  asm("v_tanh_f32 %0, %1\n\tv_nop" : "=v"(r) : "v"(x));
  return r;
#endif
}

// sigmoid(x) = 0.5 + 0.5*tanh(0.5*x): one TRANS op, branch-free.
__device__ __forceinline__ float fast_sig(float x) {
  return fmaf(0.5f, fast_tanh(0.5f * x), 0.5f);
}

// One wave32 per block; each wave owns 16 batch rows for all 1024 steps.
// gates(16x128) = [x|pad] @ WihT + h @ WhhT + b  via chained f16 WMMAs (f32 acc).
__global__ __launch_bounds__(32)
void lstm_wave_kernel(const float* __restrict__ x,
                      const float* __restrict__ Wih,
                      const float* __restrict__ Whh,
                      const float* __restrict__ bih,
                      const float* __restrict__ bhh,
                      const float* __restrict__ Wlin,
                      const float* __restrict__ blin,
                      float* __restrict__ out) {
  // tiles 0..7: W_hh B-fragments, tiles 8..15: W_ih (K-padded) B-fragments.
  // Stored per-lane in exact B-operand register order (16 f16 = 32B, 16B aligned).
  __shared__ _Float16 wstage[16][32][16];
  // row-major h staging for D-layout -> A-layout rebuild; pitch 48 halfs (96B, 16B aligned rows)
  __shared__ _Float16 hstage[16][48];

  const int lane = threadIdx.x;
  const int lo   = lane & 15;
  const int hi   = lane >> 4;
  const int b0   = blockIdx.x * 16;
  const int k0   = hi * 16;            // B-operand K range per lane half

  // ---- stage weight B-fragments as f16 ----
  #pragma unroll
  for (int nt = 0; nt < NTILE; ++nt) {
    const int n = nt * 16 + lo;        // gate column (B-matrix column = lane % 16)
    #pragma unroll
    for (int j = 0; j < 16; ++j) {
      wstage[nt][lane][j] = (_Float16)Whh[n * H_DIM + k0 + j];
      float wij = (hi == 0 && j < IN_DIM) ? Wih[n * IN_DIM + j] : 0.0f;
      wstage[8 + nt][lane][j] = (_Float16)wij;
    }
  }

  // per-lane bias (depends on N only -> broadcast across the 8 acc VGPRs)
  float bias[NTILE];
  #pragma unroll
  for (int nt = 0; nt < NTILE; ++nt)
    bias[nt] = bih[nt * 16 + lo] + bhh[nt * 16 + lo];

  // W_lin as B-fragment with only column N=0 nonzero (per-step head via WMMA)
  U16h By;
  #pragma unroll
  for (int j = 0; j < 16; ++j)
    By.h[j] = (lo == 0) ? (_Float16)Wlin[k0 + j] : (_Float16)0.0f;
  const float bl = blin[0];

  // zero h staging (t=0 reads h=0)
  for (int idx = lane; idx < 16 * 48; idx += 32)
    (&hstage[0][0])[idx] = (_Float16)0.0f;

  // cell state in D-layout: cst[ch][r] = c[M = r+8*hi][N = 16*ch + lo]
  v8f cst[2];
  #pragma unroll
  for (int r = 0; r < 8; ++r) { cst[0][r] = 0.0f; cst[1][r] = 0.0f; }

  // h_{t-1} as A-operand fragment; zero for t=0
  U16h A;
  #pragma unroll
  for (int j = 0; j < 16; ++j) A.h[j] = (_Float16)0.0f;

  for (int t = 0; t < T_LEN; ++t) {
    // ---- x_t A-fragment: row M = lane (lanes 0-15), K = 0..5, rest zero ----
    U16h ax;
    #pragma unroll
    for (int j = 0; j < 16; ++j) ax.h[j] = (_Float16)0.0f;
    if (hi == 0) {
      const float* xp = x + ((size_t)(b0 + lo) * T_LEN + t) * IN_DIM;
      #pragma unroll
      for (int j = 0; j < IN_DIM; ++j) ax.h[j] = (_Float16)xp[j];
    }

    float hn[2][8];
    #pragma unroll
    for (int ch = 0; ch < 2; ++ch) {          // column half of H (N 0-15 / 16-31)
      v8f accs[4];
      #pragma unroll
      for (int gi = 0; gi < 4; ++gi) {        // gate i, f, g, o
        const int nt = ch + 2 * gi;           // gate tile index in 0..7
        v8f acc;
        #pragma unroll
        for (int r = 0; r < 8; ++r) acc[r] = bias[nt];

        U16h bw;
        const uint4* wp = reinterpret_cast<const uint4*>(&wstage[8 + nt][lane][0]);
        bw.u[0] = wp[0]; bw.u[1] = wp[1];
        acc = __builtin_amdgcn_wmma_f32_16x16x32_f16(false, ax.v, false, bw.v,
                                                     (short)0, acc, false, false);
        U16h bh;
        const uint4* hp = reinterpret_cast<const uint4*>(&wstage[nt][lane][0]);
        bh.u[0] = hp[0]; bh.u[1] = hp[1];
        acc = __builtin_amdgcn_wmma_f32_16x16x32_f16(false, A.v, false, bh.v,
                                                     (short)0, acc, false, false);
        accs[gi] = acc;
      }
      // pointwise LSTM update, purely elementwise in D-layout, branch-free:
      // 5 hardware v_tanh per element, no exp/rcp/clamping sequences.
      #pragma unroll
      for (int r = 0; r < 8; ++r) {
        float gI = accs[0][r], gF = accs[1][r], gG = accs[2][r], gO = accs[3][r];
        float cc = fast_sig(gF) * cst[ch][r] + fast_sig(gI) * fast_tanh(gG);
        cst[ch][r] = cc;
        hn[ch][r]  = fast_sig(gO) * fast_tanh(cc);
      }
    }

    // ---- stage new h (row-major [M][K]) and rebuild A-operand fragment ----
    #pragma unroll
    for (int ch = 0; ch < 2; ++ch)
      #pragma unroll
      for (int r = 0; r < 8; ++r)
        hstage[r + 8 * hi][lo + 16 * ch] = (_Float16)hn[ch][r];

    asm volatile("s_wait_dscnt 0" ::: "memory");  // same-wave DS store->load ordering

    const uint4* hrow = reinterpret_cast<const uint4*>(&hstage[lo][0]);
    A.u[0] = hrow[2 * hi];     // lanes<16: K0..7,  lanes>=16: K16..23
    A.u[1] = hrow[2 * hi + 1]; // lanes<16: K8..15, lanes>=16: K24..31

    // ---- per-step head: y = h_t @ Wlin^T + b via one WMMA (column 0 of D) ----
    v8f cy;
    #pragma unroll
    for (int r = 0; r < 8; ++r) cy[r] = bl;
    cy = __builtin_amdgcn_wmma_f32_16x16x32_f16(false, A.v, false, By.v,
                                                (short)0, cy, false, false);
    if (lo == 0) {
      const int mb = b0 + 8 * hi;
      #pragma unroll
      for (int r = 0; r < 8; ++r)
        out[(size_t)(mb + r) * T_LEN + t] = cy[r];
    }
  }
}

extern "C" void kernel_launch(void* const* d_in, const int* in_sizes, int n_in,
                              void* d_out, int out_size, void* d_ws, size_t ws_size,
                              hipStream_t stream) {
  (void)in_sizes; (void)n_in; (void)d_ws; (void)ws_size; (void)out_size;
  const float* x    = (const float*)d_in[0];
  const float* Wih  = (const float*)d_in[1];
  const float* Whh  = (const float*)d_in[2];
  const float* bih  = (const float*)d_in[3];
  const float* bhh  = (const float*)d_in[4];
  const float* Wlin = (const float*)d_in[5];
  const float* blin = (const float*)d_in[6];
  float* out = (float*)d_out;

  dim3 grid(B_TOT / 16);  // 256 waves, one 16-row batch tile each
  dim3 block(32);         // exactly one wave32 -> EXEC all-ones for WMMA
  lstm_wave_kernel<<<grid, block, 0, stream>>>(x, Wih, Whh, bih, bhh, Wlin, blin, out);
}
